// GCNLayer_HSI_spectral_36206574305749
// MI455X (gfx1250) — matrix operand order repaired and checked by reading, not compiled, not run on hardware
//
#include <hip/hip_runtime.h>
#include <math.h>

#define N_NODES 8192
#define D 256
#define K_TOP 5

typedef __attribute__((ext_vector_type(16))) _Float16 v16h;
typedef __attribute__((ext_vector_type(8)))  _Float16 v8h;
typedef __attribute__((ext_vector_type(8)))  float    v8f;

// ---- WMMA fragment loaders (CDNA5 16x16x32 f16 layouts, wave32) ----
// A (16x32, MxK): lane L holds row M=L%16; half e -> K = 16*(L/16) + e
__device__ __forceinline__ v16h load_a_frag(const _Float16* __restrict__ base, int rowBase,
                                            int lda, int kb, int lane) {
  const _Float16* p = base + (size_t)(rowBase + (lane & 15)) * lda + kb + ((lane >> 4) << 4);
  v8h x0 = *(const v8h*)(p);
  v8h x1 = *(const v8h*)(p + 8);
  v16h a;
#pragma unroll
  for (int i = 0; i < 8; ++i) { a[i] = x0[i]; a[i + 8] = x1[i]; }
  return a;
}

// B (32x16, KxN): lane L holds col N=L%16; half e -> K = 16*(L/16) + e.
// 'base' is the N-major matrix, so the 16 halves are contiguous.
__device__ __forceinline__ v16h load_b_frag(const _Float16* __restrict__ base, int colBase,
                                            int ldb, int kb, int lane) {
  const _Float16* p = base + (size_t)(colBase + (lane & 15)) * ldb + kb + ((lane >> 4) << 4);
  return *(const v16h*)p;
}

// ---- 1. BatchNorm batch stats -> per-column scale/shift ----
__global__ void bn_stats_kernel(const float* __restrict__ H, const float* __restrict__ gamma,
                                const float* __restrict__ beta, float* __restrict__ scale,
                                float* __restrict__ shift) {
  __shared__ float s_sum[256];
  __shared__ float s_sq[256];
  int d = blockIdx.x, t = threadIdx.x;
  float s = 0.f, q = 0.f;
  for (int n = t; n < N_NODES; n += 256) {
    float x = H[(size_t)n * D + d];
    s += x; q += x * x;
  }
  s_sum[t] = s; s_sq[t] = q;
  __syncthreads();
  for (int off = 128; off > 0; off >>= 1) {
    if (t < off) { s_sum[t] += s_sum[t + off]; s_sq[t] += s_sq[t + off]; }
    __syncthreads();
  }
  if (t == 0) {
    float mu = s_sum[0] * (1.f / N_NODES);
    float var = s_sq[0] * (1.f / N_NODES) - mu * mu;   // biased variance
    float sc = gamma[d] * rsqrtf(var + 1e-5f);
    scale[d] = sc;
    shift[d] = beta[d] - mu * sc;
  }
}

// ---- 2. Apply BN, convert to f16 ----
__global__ void bn_apply_kernel(const float* __restrict__ H, const float* __restrict__ scale,
                                const float* __restrict__ shift, _Float16* __restrict__ Hn) {
  int idx = blockIdx.x * 256 + threadIdx.x;       // N*D / 256 blocks, exact
  int d = idx & (D - 1);
  Hn[idx] = (_Float16)(H[idx] * scale[d] + shift[d]);
}

// ---- 3. Transpose + convert weights to f16 (N-major for B-fragments) ----
__global__ void wt_convert_kernel(const float* __restrict__ Wth, const float* __restrict__ Wout,
                                  _Float16* __restrict__ Wth_t, _Float16* __restrict__ Wout_t) {
  int n = blockIdx.x, k = threadIdx.x;
  const float* W = blockIdx.y ? Wout : Wth;
  _Float16* Wt = blockIdx.y ? Wout_t : Wth_t;
  Wt[n * D + k] = (_Float16)(W[k * D + n]);
}

// ---- 4. Dual GEMM: Hx = Hn@W_theta + b_theta (y=0), HWo = Hn@W_out + b_out (y=1) ----
// One-step software pipeline: two ping-pong B fragment registers; the loads for
// step s+1 are issued before the WMMA of step s. 8 independent accumulators
// (one per 16-col strip) keep the WMMAs pipelineable; live set ~160 VGPRs.
__global__ void gemm_hn_w_kernel(const _Float16* __restrict__ Hn, const _Float16* __restrict__ Wth_t,
                                 const _Float16* __restrict__ Wout_t, const float* __restrict__ b_th,
                                 const float* __restrict__ b_o, float* __restrict__ Hx,
                                 float* __restrict__ HWo) {
  const _Float16* Wt = blockIdx.y ? Wout_t : Wth_t;
  const float* bias = blockIdx.y ? b_o : b_th;
  float* Out = blockIdx.y ? HWo : Hx;
  int w = threadIdx.x >> 5, lane = threadIdx.x & 31;
  int rowBase = blockIdx.x * 64 + (w & 3) * 16;
  int colGroup = (w >> 2) * 128;

  v16h a[8];
#pragma unroll
  for (int kk = 0; kk < 8; ++kk) a[kk] = load_a_frag(Hn, rowBase, D, kk * 32, lane);

  v8f cacc[8];
#pragma unroll
  for (int cs = 0; cs < 8; ++cs) cacc[cs] = (v8f){};

  v16h bb0 = load_b_frag(Wt, colGroup, D, 0, lane);   // step 0 fragment
  v16h bb1;
#pragma unroll
  for (int s = 0; s < 64; ++s) {                      // s = cs*8 + kk
    int cs = s >> 3, kk = s & 7;
    if (s + 1 < 64) {                                 // prefetch step s+1
      int ns = s + 1;
      if (s & 1) bb0 = load_b_frag(Wt, colGroup + (ns >> 3) * 16, D, (ns & 7) * 32, lane);
      else       bb1 = load_b_frag(Wt, colGroup + (ns >> 3) * 16, D, (ns & 7) * 32, lane);
    }
    v16h bcur = (s & 1) ? bb1 : bb0;
    cacc[cs] = __builtin_amdgcn_wmma_f32_16x16x32_f16(false, a[kk], false, bcur,
                                                      (short)0, cacc[cs], false, false);
  }

#pragma unroll
  for (int cs = 0; cs < 8; ++cs) {
    int col = colGroup + cs * 16 + (lane & 15);
    float bv = bias[col];
#pragma unroll
    for (int v = 0; v < 8; ++v) {
      int row = rowBase + v + ((lane >> 4) << 3);
      Out[(size_t)row * D + col] = cacc[cs][v] + bv;
    }
  }
}

// ---- 5. Row L2-normalize Hx -> F1 (f16), one wave per row ----
__global__ void rownorm_kernel(const float* __restrict__ Hx, _Float16* __restrict__ F1) {
  int w = threadIdx.x >> 5, lane = threadIdx.x & 31;
  int row = blockIdx.x * 8 + w;
  const float* p = Hx + (size_t)row * D;
  float s = 0.f;
#pragma unroll
  for (int i = 0; i < 8; ++i) { float x = p[lane + i * 32]; s += x * x; }
#pragma unroll
  for (int m = 16; m > 0; m >>= 1) s += __shfl_xor(s, m, 32);
  float inv = 1.f / fmaxf(sqrtf(s), 1e-12f);
  _Float16* q = F1 + (size_t)row * D;
#pragma unroll
  for (int i = 0; i < 8; ++i) q[lane + i * 32] = (_Float16)(p[lane + i * 32] * inv);
}

// ---- 6. Similarity (F1@F1^T via WMMA) + streaming per-row top-5 ----
// Column tiles (64x256 f16 = 32KB) are staged into LDS with CDNA5 async
// global->LDS copies (ASYNCcnt), double-buffered, pre-swizzled into
// B-fragment order so ds_load_b128 reads are conflict-free.
// LDS fragment store layout: [kk(8)][g4(4)][plane(2)][lane(32)][8 halves]
//   plane0 = fragment halves e=0..7, plane1 = e=8..15 for that lane.
#define SB_HALF_PER_BUF (8 * 4 * 2 * 32 * 8)   // 16384 halves = 32KB

__device__ __forceinline__ void issue_tile_async(const _Float16* __restrict__ F1, int colBase,
                                                 _Float16* bufBase, int w, int lane) {
#pragma unroll
  for (int i = 0; i < 8; ++i) {
    int idx = w * 8 + i;                 // 0..63 units per wave-instr, x32 lanes
    int kk = idx & 7;
    int g4 = (idx >> 3) & 3;
    int plane = idx >> 5;
    int row = colBase + g4 * 16 + (lane & 15);
    int khalf = kk * 32 + ((lane >> 4) << 4) + plane * 8;
    const _Float16* gp = F1 + (size_t)row * D + khalf;                       // 16B chunk
    _Float16* lp = bufBase + (((kk * 4 + g4) * 2 + plane) * 32 + lane) * 8;  // 16B slot
    unsigned lds = (unsigned)(uintptr_t)lp;                 // flat LDS addr low bits = LDS offset
    unsigned long long ga = (unsigned long long)(uintptr_t)gp;
    asm volatile("global_load_async_to_lds_b128 %0, %1, off"
                 :: "v"(lds), "v"(ga) : "memory");
  }
}

__device__ __forceinline__ v16h load_b_lds(const _Float16* bufBase, int kk, int scol, int lane) {
  const _Float16* lo = bufBase + (((kk * 4 + scol) * 2 + 0) * 32 + lane) * 8;
  const _Float16* hi = bufBase + (((kk * 4 + scol) * 2 + 1) * 32 + lane) * 8;
  v8h x0 = *(const v8h*)lo;
  v8h x1 = *(const v8h*)hi;
  v16h b;
#pragma unroll
  for (int i = 0; i < 8; ++i) { b[i] = x0[i]; b[i + 8] = x1[i]; }
  return b;
}

__global__ void sim_topk_kernel(const _Float16* __restrict__ F1, float* __restrict__ topval,
                                int* __restrict__ topidx, float* __restrict__ rowsum) {
  __shared__ __align__(16) _Float16 sB[2][SB_HALF_PER_BUF];   // 64KB double buffer
  __shared__ float sS[64 * 68];                               // S tile (reused for merge)
  int w = threadIdx.x >> 5, lane = threadIdx.x & 31;
  int t = threadIdx.x;
  int rBase = blockIdx.x * 64;
  int rowSub = (w & 3) * 16;                 // local row base of this wave
  int scol0 = (w >> 2) * 2;                  // first of 2 local col subtiles (units of 16)

  v16h a[8];                                 // rows held in VGPRs for all col tiles
#pragma unroll
  for (int kk = 0; kk < 8; ++kk) a[kk] = load_a_frag(F1, rBase + rowSub, D, kk * 32, lane);

  // streaming top-5: thread t covers row (t&63), column quadrant (t>>6)
  int myRow = t & 63;
  int myQ = t >> 6;
  float t0 = -2.f, t1 = -2.f, t2 = -2.f, t3 = -2.f, t4 = -2.f;
  int i0 = 0, i1 = 0, i2 = 0, i3 = 0, i4 = 0;

  issue_tile_async(F1, 0, sB[0], w, lane);   // prologue: tile 0

  const int nTiles = N_NODES / 64;
  for (int ct = 0; ct < nTiles; ++ct) {
    const _Float16* buf = sB[ct & 1];
    if (ct + 1 < nTiles) {
      issue_tile_async(F1, (ct + 1) * 64, sB[(ct + 1) & 1], w, lane);
      asm volatile("s_wait_asynccnt 0x8" ::: "memory");   // tile ct done; ct+1 in flight
    } else {
      asm volatile("s_wait_asynccnt 0x0" ::: "memory");
    }
    __syncthreads();                                       // tile ct visible to all waves

    v8f c0 = {}, c1 = {};
#pragma unroll
    for (int kk = 0; kk < 8; ++kk) {
      v16h b0 = load_b_lds(buf, kk, scol0, lane);
      v16h b1 = load_b_lds(buf, kk, scol0 + 1, lane);
      c0 = __builtin_amdgcn_wmma_f32_16x16x32_f16(false, a[kk], false, b0, (short)0, c0, false, false);
      c1 = __builtin_amdgcn_wmma_f32_16x16x32_f16(false, a[kk], false, b1, (short)0, c1, false, false);
    }
    int lr = rowSub + ((lane >> 4) << 3);
    int lc = scol0 * 16 + (lane & 15);
#pragma unroll
    for (int v = 0; v < 8; ++v) {
      sS[(lr + v) * 68 + lc] = c0[v];
      sS[(lr + v) * 68 + lc + 16] = c1[v];
    }
    __syncthreads();

    {
      const float* rowS = &sS[myRow * 68 + myQ * 16];
      int idxBase = ct * 64 + myQ * 16;
      for (int j = 0; j < 16; ++j) {
        float v = rowS[j];
        int idx = idxBase + j;
        if (v > t4) {
          if (v > t2) {
            if (v > t1) {
              if (v > t0) { t4=t3;i4=i3; t3=t2;i3=i2; t2=t1;i2=i1; t1=t0;i1=i0; t0=v;i0=idx; }
              else        { t4=t3;i4=i3; t3=t2;i3=i2; t2=t1;i2=i1; t1=v;i1=idx; }
            } else        { t4=t3;i4=i3; t3=t2;i3=i2; t2=v;i2=idx; }
          } else {
            if (v > t3)   { t4=t3;i4=i3; t3=v;i3=idx; }
            else          { t4=v;i4=idx; }
          }
        }
      }
    }
    __syncthreads();                                       // protects sS and buf reuse
  }

  // merge 4 partial top-5 lists per row (reuse sS region)
  float* mv = sS;                          // 64*20 floats
  int*   mi = (int*)(sS + 64 * 20);        // 64*20 ints
  {
    float pv[5] = { t0, t1, t2, t3, t4 };
    int   pi[5] = { i0, i1, i2, i3, i4 };
#pragma unroll
    for (int k = 0; k < K_TOP; ++k) {
      mv[myRow * 20 + myQ * 5 + k] = pv[k];
      mi[myRow * 20 + myQ * 5 + k] = pi[k];
    }
  }
  __syncthreads();

  if (t < 64) {
    float bv[5] = { -3.f, -3.f, -3.f, -3.f, -3.f };
    int   bi[5] = { 0x7fffffff, 0x7fffffff, 0x7fffffff, 0x7fffffff, 0x7fffffff };
    for (int c = 0; c < 20; ++c) {
      float v = mv[t * 20 + c];
      int  id = mi[t * 20 + c];
#pragma unroll
      for (int k = 0; k < 5; ++k) {
        bool better = (v > bv[k]) || (v == bv[k] && id < bi[k]);   // jax tie-break
        if (better) {
#pragma unroll
          for (int m = 4; m > k; --m) { bv[m] = bv[m - 1]; bi[m] = bi[m - 1]; }
          bv[k] = v; bi[k] = id;
          break;
        }
      }
    }
    int row = rBase + t;
    float s = 0.f;
#pragma unroll
    for (int k = 0; k < K_TOP; ++k) {
      float d = fminf(fmaxf(bv[k], -1.f), 1.f);
      float sam = acosf(d);
      if (isnan(sam)) sam = 0.f;                      // nan_to_num
      float e = expf(-0.2f * sam);
      float aV = 1.f / (1.f + expf(-e));              // sigmoid
      aV = fmaxf(aV, 0.1f);
      s += aV;
      topval[row * K_TOP + k] = aV;
      topidx[row * K_TOP + k] = bi[k];
    }
    rowsum[row] = s;
  }
}

// ---- 7. Zero the dense A output region (256 MB) ----
__global__ void zero_a_kernel(float4* __restrict__ A4) {
  const int n4 = N_NODES * (N_NODES / 4);
  for (int i = blockIdx.x * blockDim.x + threadIdx.x; i < n4; i += gridDim.x * blockDim.x)
    A4[i] = make_float4(0.f, 0.f, 0.f, 0.f);
}

// ---- 8. Scatter the 5 kept affinity values per row ----
__global__ void scatter_a_kernel(const float* __restrict__ topval, const int* __restrict__ topidx,
                                 float* __restrict__ A) {
  int e = blockIdx.x * 256 + threadIdx.x;
  if (e >= N_NODES * K_TOP) return;
  int i = e / K_TOP;
  A[(size_t)i * N_NODES + topidx[e]] = topval[e];
}

// ---- 9. out = LeakyReLU(A_hat @ HWo), 5-nnz/row gather SpMM ----
__global__ void out_kernel(const float* __restrict__ HWo, const float* __restrict__ topval,
                           const int* __restrict__ topidx, const float* __restrict__ rowsum,
                           float* __restrict__ out) {
  __shared__ float sw[K_TOP];
  __shared__ int sj[K_TOP];
  int i = blockIdx.x, t = threadIdx.x;
  if (t < K_TOP) {
    int j = topidx[i * K_TOP + t];
    sw[t] = rsqrtf(rowsum[i]) * topval[i * K_TOP + t] * rsqrtf(rowsum[j]);
    sj[t] = j;
  }
  __syncthreads();
  float acc = 0.f;
#pragma unroll
  for (int k = 0; k < K_TOP; ++k) acc += sw[k] * HWo[(size_t)sj[k] * D + t];
  out[(size_t)i * D + t] = acc >= 0.f ? acc : 0.01f * acc;
}

extern "C" void kernel_launch(void* const* d_in, const int* in_sizes, int n_in,
                              void* d_out, int out_size, void* d_ws, size_t ws_size,
                              hipStream_t stream) {
  (void)in_sizes; (void)n_in; (void)out_size; (void)ws_size;
  const float* H       = (const float*)d_in[0];
  const float* bn_g    = (const float*)d_in[1];
  const float* bn_b    = (const float*)d_in[2];
  const float* W_theta = (const float*)d_in[3];
  const float* b_theta = (const float*)d_in[4];
  const float* W_out   = (const float*)d_in[5];
  const float* b_out   = (const float*)d_in[6];

  char* ws = (char*)d_ws;
  float*    scale  = (float*)(ws + 0);                       //   1 KB
  float*    shift  = (float*)(ws + 1024);                    //   1 KB
  float*    rowsum = (float*)(ws + 2048);                    //  32 KB
  float*    topval = (float*)(ws + 34816);                   // 160 KB
  int*      topidx = (int*)  (ws + 198656);                  // 160 KB
  _Float16* Hn_h   = (_Float16*)(ws + 362496);               //   4 MB
  _Float16* Wth_t  = (_Float16*)(ws + 4556800);              // 128 KB
  _Float16* Wout_t = (_Float16*)(ws + 4687872);              // 128 KB
  _Float16* F1_h   = (_Float16*)(ws + 4818944);              //   4 MB
  float*    Hx     = (float*)(ws + 9013248);                 //   8 MB
  float*    HWo    = (float*)(ws + 17401856);                //   8 MB  (end ~24.6 MB)

  float* out_f = (float*)d_out;                              // [8192 x 256]
  float* A_out = out_f + (size_t)N_NODES * D;                // [8192 x 8192]

  bn_stats_kernel<<<D, 256, 0, stream>>>(H, bn_g, bn_b, scale, shift);
  bn_apply_kernel<<<(N_NODES * D) / 256, 256, 0, stream>>>(H, scale, shift, Hn_h);
  wt_convert_kernel<<<dim3(D, 2), D, 0, stream>>>(W_theta, W_out, Wth_t, Wout_t);
  gemm_hn_w_kernel<<<dim3(N_NODES / 64, 2), 256, 0, stream>>>(Hn_h, Wth_t, Wout_t,
                                                              b_theta, b_out, Hx, HWo);
  rownorm_kernel<<<N_NODES / 8, 256, 0, stream>>>(Hx, F1_h);
  sim_topk_kernel<<<N_NODES / 64, 256, 0, stream>>>(F1_h, topval, topidx, rowsum);
  zero_a_kernel<<<16384, 256, 0, stream>>>((float4*)A_out);
  scatter_a_kernel<<<(N_NODES * K_TOP + 255) / 256, 256, 0, stream>>>(topval, topidx, A_out);
  out_kernel<<<N_NODES, D, 0, stream>>>(HWo, topval, topidx, rowsum, out_f);
}
